// CRsAE1D_16552803958953
// MI455X (gfx1250) — compile-verified, matrix operand order checked by reading, not run on hardware
//
#include <hip/hip_runtime.h>
#include <math.h>

// Problem constants (from reference)
#define BB 32
#define NN 16384
#define CC 64
#define KK 21
#define MM (NN - KK + 1)      // 16364 (enc_dim)
#define MP NN                 // padded M for buffer layout
#define TT 30
#define INV_L      0.1f       // 1/L
#define LAM_OVER_L 0.01f      // LAM/L

// Tiling
#define TM 128                            // m-positions per block (iter kernel)
#define WH (TM + 2*(KK-1) + 8)            // 176: halo width, multiple of 16
#define RW (TM + KK - 1)                  // 148: residual width
#define TN 128                            // n-positions per block (final kernel)
#define WF 160                            // final halo width (148 -> 160)

#define TILE_HALVES (WH * CC)             // 11264 f16 elements = 22528 bytes

typedef __attribute__((ext_vector_type(16))) _Float16     v16h;
typedef __attribute__((ext_vector_type(8)))  _Float16     v8h;
typedef __attribute__((ext_vector_type(8)))  float        v8f;
typedef __attribute__((ext_vector_type(4)))  int          v4i;
typedef __attribute__((ext_vector_type(8)))  int          v8i;
typedef __attribute__((ext_vector_type(4)))  unsigned int v4u;

#if __has_builtin(__builtin_amdgcn_tensor_load_to_lds)
#define HAVE_TDM 1
#endif
#if __has_builtin(__builtin_amdgcn_global_load_async_to_lds_b128)
#define HAVE_ASYNC_LDS 1
typedef __attribute__((address_space(1))) v4i GV4;   // global int4
typedef __attribute__((address_space(3))) v4i LV4;   // LDS int4
#endif

// wave32 16-bit A/B fragment K-index mapping for V_WMMA_*_16X16X32:
// lanes 0-15 hold K in {0..7, 16..23}, lanes 16-31 hold K in {8..15, 24..31}
__device__ __forceinline__ int kmap(int lane, int e) {
  return ((e >> 3) << 4) | ((lane >> 4) << 3) | (e & 7);
}

// Load a v16h fragment whose 16 elements are two contiguous 8-half chunks
// at p and p+16 halves (16-byte aligned) -> 2x ds_load_b128.
__device__ __forceinline__ v16h ld_frag(const _Float16* p) {
  union { v16h v; v8h h[2]; } u;
  u.h[0] = *(const v8h*)(p);
  u.h[1] = *(const v8h*)(p + 16);
  return u.v;
}

__device__ __forceinline__ v8f wmma_f16(v16h a, v16h b, v8f c) {
  return __builtin_amdgcn_wmma_f32_16x16x32_f16(false, a, false, b,
                                                (short)0, c, false, false);
}

#ifdef HAVE_TDM
// Issue a single TDM linear copy: nhalves f16 from gsrc -> LDS at ldst.
// D# per CDNA5 ISA ch.8: 2-D tensor (dim1=1), data_size=2B, tile==tensor.
__device__ __forceinline__ void tdm_copy_to_lds(const _Float16* gsrc, void* ldst,
                                                unsigned int nhalves) {
  unsigned long long ga = (unsigned long long)(const void*)gsrc;
  unsigned int lo = (unsigned int)(unsigned long long)(size_t)ldst; // LDS byte addr
  v4u g0;
  g0.x = 1u;                                            // count=1, valid user D#
  g0.y = lo;                                            // lds_addr
  g0.z = (unsigned int)ga;                              // global_addr[31:0]
  g0.w = ((unsigned int)(ga >> 32) & 0x01FFFFFFu) | (2u << 30); // addr[56:32], type=2
  v8i g1;
  g1[0] = (int)(1u << 16);                              // data_size=1 (2 bytes/elem)
  g1[1] = (int)((nhalves & 0xFFFFu) << 16);             // tensor_dim0[15:0]
  g1[2] = (int)(((nhalves >> 16) & 0xFFFFu) | (1u << 16)); // dim0 hi, tensor_dim1=1
  g1[3] = (int)((nhalves & 0xFFFFu) << 16);             // tile_dim0 (== tensor_dim0)
  g1[4] = 1;                                            // tile_dim1=1
  g1[5] = (int)nhalves;                                 // tensor_dim0_stride
  g1[6] = 0; g1[7] = 0;
  v4i z = {0, 0, 0, 0};
#if __clang_major__ >= 23
  v8i z8 = {0, 0, 0, 0, 0, 0, 0, 0};
  __builtin_amdgcn_tensor_load_to_lds(g0, g1, z, z, z8, 0);
#else
  __builtin_amdgcn_tensor_load_to_lds(g0, g1, z, z, 0);
#endif
}
#endif

__global__ void crsae_init(float* __restrict__ XO, _Float16* __restrict__ XT0, size_t n) {
  size_t i = (size_t)blockIdx.x * blockDim.x + threadIdx.x;
  size_t stride = (size_t)gridDim.x * blockDim.x;
  for (; i < n; i += stride) { XO[i] = 0.0f; XT0[i] = (_Float16)0.0f; }
}

// One fused FISTA iteration. Block = (m-chunk, batch). Buffers [b][m][c], c fastest.
__global__ __launch_bounds__(256)
void crsae_iter(const float* __restrict__ y, const float* __restrict__ H,
                const _Float16* __restrict__ XTr,   // x_tmp (read, f16)
                float* __restrict__ XO,             // x_old in / x_new out (f32)
                _Float16* __restrict__ XTw,         // x_tmp next (write, f16)
                float mom)
{
  __shared__ __align__(16) _Float16 sHT[32 * CC];   // H^T [k=32][c=64], rows k>=21 zero
  __shared__ __align__(16) _Float16 sH2[CC * 32];   // H   [c=64][k=32], cols k>=21 zero
  __shared__ __align__(16) _Float16 sXT[TILE_HALVES]; // x_tmp halo tile [u][c]
  __shared__ __align__(16) float    sZ [32 * WH];   // Z = H^T X   [k][u]
  __shared__ __align__(16) _Float16 sRes[WH];       // residual (f16 for WMMA A)

  const int tid  = threadIdx.x;
  const int lane = tid & 31;
  const int wid  = tid >> 5;
  const int g8   = (lane >> 4) << 3;
  const int l16  = lane & 15;
  const int b    = blockIdx.y;
  const int m0   = blockIdx.x * TM;
  const size_t xbase = (size_t)b * MP * CC;

  // ---- stage H in both orientations (f16, zero-padded in k) ----
  for (int i = tid; i < 32 * CC; i += 256) {
    int k = i >> 6, c = i & 63;
    _Float16 v = (k < KK) ? (_Float16)H[c * KK + k] : (_Float16)0.0f;
    sHT[i] = v;                 // [k][c]
    sH2[c * 32 + k] = v;        // [c][k]
  }

  // ---- stage x_tmp halo tile: u in [0, WH), m = m0-(K-1)+u, layout [u][c] ----
  const bool interior = (m0 >= (KK - 1)) && (m0 - (KK - 1) + WH <= MM);
  if (interior) {
    _Float16* gsrc = const_cast<_Float16*>(XTr) + xbase + (size_t)(m0 - (KK - 1)) * CC;
#if defined(HAVE_TDM)
    if (wid == 0) {                       // one wave issues the DMA descriptor
      tdm_copy_to_lds(gsrc, &sXT[0], (unsigned int)TILE_HALVES);
      __builtin_amdgcn_s_wait_tensorcnt(0);
    }
#elif defined(HAVE_ASYNC_LDS)
    for (int i = tid * 8; i < TILE_HALVES; i += 256 * 8) {
      __builtin_amdgcn_global_load_async_to_lds_b128(
          (GV4*)(gsrc + i), (LV4*)&sXT[i], 0, 0);
    }
    asm volatile("s_wait_asynccnt 0x0" ::: "memory");
#else
    for (int i = tid; i < TILE_HALVES; i += 256) sXT[i] = gsrc[i];
#endif
  } else {
    for (int i = tid; i < TILE_HALVES; i += 256) {
      int u = i >> 6, c = i & 63;
      int m = m0 - (KK - 1) + u;
      sXT[i] = (m >= 0 && m < MM) ? XTr[xbase + (size_t)m * CC + c] : (_Float16)0.0f;
    }
  }
  __syncthreads();

  // ---- Phase B: Z[k,u] = sum_c H[c,k] * x_tmp[c,u]  (WMMA, K-dim = c) ----
  {
    v16h aF[2][2];
#pragma unroll
    for (int kt = 0; kt < 2; ++kt)
#pragma unroll
      for (int c2 = 0; c2 < 2; ++c2)
        aF[kt][c2] = ld_frag(&sHT[(kt * 16 + l16) * CC + c2 * 32 + g8]);

    const int NT = WH / 16;                 // 11 u-tiles
    for (int nt = wid; nt < NT; nt += 8) {
      int u0 = nt * 16;
      v16h b0 = ld_frag(&sXT[(u0 + l16) * CC + g8]);
      v16h b1 = ld_frag(&sXT[(u0 + l16) * CC + 32 + g8]);
#pragma unroll
      for (int kt = 0; kt < 2; ++kt) {
        v8f acc = {};
        acc = wmma_f16(aF[kt][0], b0, acc);
        acc = wmma_f16(aF[kt][1], b1, acc);
#pragma unroll
        for (int r = 0; r < 8; ++r)
          sZ[(kt * 16 + r + g8) * WH + u0 + l16] = acc[r];
      }
    }
  }
  __syncthreads();

  // ---- residual: res[t] = y[m0+t] - sum_k Z[k][t+(K-1)-k],  t in [0,RW) ----
  for (int t = tid; t < WH; t += 256) {
    float v = 0.0f;
    if (t < RW) {
      int n = m0 + t;
      float a = (n < NN) ? y[(size_t)b * NN + n] : 0.0f;
#pragma unroll
      for (int k = 0; k < KK; ++k) a -= sZ[k * WH + (t + (KK - 1) - k)];
      v = a;
    }
    sRes[t] = (_Float16)v;
  }
  __syncthreads();

  // ---- Phase C: grad[m,c] = sum_k res[m+k] H[c,k]; shrink + momentum ----
  {
    const int mt = wid;                     // 8 waves <-> 8 m-tiles
    v16h aR;
#pragma unroll
    for (int e = 0; e < 16; ++e)
      aR[e] = sRes[mt * 16 + l16 + kmap(lane, e)];  // per-lane unaligned window

#pragma unroll
    for (int ct = 0; ct < 4; ++ct) {
      v16h bH = ld_frag(&sH2[(ct * 16 + l16) * 32 + g8]);
      v8f acc = {};
      acc = wmma_f16(aR, bH, acc);
      int c = ct * 16 + l16;
#pragma unroll
      for (int r = 0; r < 8; ++r) {
        int m = m0 + mt * 16 + r + g8;
        if (m < MM) {
          size_t idx = xbase + (size_t)m * CC + c;
          float xt = (float)XTr[idx];
          float xo = XO[idx];
          float xn = xt + acc[r] * INV_L - LAM_OVER_L;  // relu(x_tmp + conv/L - lam/L)
          xn = xn > 0.0f ? xn : 0.0f;
          XO[idx]  = xn;                                // becomes next x_old
          XTw[idx] = (_Float16)(xn + mom * (xn - xo));  // x_tmp next
        }
      }
    }
  }
}

// Final: y_hat = convT(x_new); also transpose-copy x_new into [B,C,M] output.
__global__ __launch_bounds__(256)
void crsae_final(const float* __restrict__ H, const float* __restrict__ XO,
                 float* __restrict__ out)
{
  __shared__ __align__(16) _Float16 sHT[32 * CC];
  __shared__ __align__(16) _Float16 sXN[WF * CC];   // [u][c]
  __shared__ __align__(16) float    sZ [32 * WF];

  const int tid  = threadIdx.x;
  const int lane = tid & 31;
  const int wid  = tid >> 5;
  const int g8   = (lane >> 4) << 3;
  const int l16  = lane & 15;
  const int b    = blockIdx.y;
  const int n0   = blockIdx.x * TN;
  const size_t xbase = (size_t)b * MP * CC;

  for (int i = tid; i < 32 * CC; i += 256) {
    int k = i >> 6, c = i & 63;
    sHT[i] = (k < KK) ? (_Float16)H[c * KK + k] : (_Float16)0.0f;
  }
  for (int i = tid; i < WF * CC; i += 256) {
    int u = i >> 6, c = i & 63;
    int m = n0 - (KK - 1) + u;
    float v = 0.0f;
    if (m >= 0 && m < MM) v = XO[xbase + (size_t)m * CC + c];
    sXN[i] = (_Float16)v;
  }
  __syncthreads();

  {
    v16h aF[2][2];
#pragma unroll
    for (int kt = 0; kt < 2; ++kt)
#pragma unroll
      for (int c2 = 0; c2 < 2; ++c2)
        aF[kt][c2] = ld_frag(&sHT[(kt * 16 + l16) * CC + c2 * 32 + g8]);

    const int NT = WF / 16;                 // 10 u-tiles
    for (int nt = wid; nt < NT; nt += 8) {
      int u0 = nt * 16;
      v16h b0 = ld_frag(&sXN[(u0 + l16) * CC + g8]);
      v16h b1 = ld_frag(&sXN[(u0 + l16) * CC + 32 + g8]);
#pragma unroll
      for (int kt = 0; kt < 2; ++kt) {
        v8f acc = {};
        acc = wmma_f16(aF[kt][0], b0, acc);
        acc = wmma_f16(aF[kt][1], b1, acc);
#pragma unroll
        for (int r = 0; r < 8; ++r)
          sZ[(kt * 16 + r + g8) * WF + u0 + l16] = acc[r];
      }
    }
  }
  __syncthreads();

  // y_hat[n0+t] = sum_k Z[k][t+(K-1)-k]
  for (int t = tid; t < TN; t += 256) {
    float a = 0.0f;
#pragma unroll
    for (int k = 0; k < KK; ++k) a += sZ[k * WF + (t + (KK - 1) - k)];
    out[(size_t)b * NN + n0 + t] = a;
  }
  // transpose-copy x_new: [b][m][c] (ws) -> [b][c][m] (d_out)
  const size_t OUT_X = (size_t)BB * NN;
  for (int i = tid; i < TN * CC; i += 256) {
    int mloc = i >> 6, c = i & 63;
    int m = n0 + mloc;
    if (m < MM)
      out[OUT_X + ((size_t)b * CC + c) * MM + m] = XO[xbase + (size_t)m * CC + c];
  }
}

extern "C" void kernel_launch(void* const* d_in, const int* in_sizes, int n_in,
                              void* d_out, int out_size, void* d_ws, size_t ws_size,
                              hipStream_t stream) {
  (void)in_sizes; (void)n_in; (void)out_size; (void)ws_size;
  const float* y = (const float*)d_in[0];
  const float* H = (const float*)d_in[1];
  float* out = (float*)d_out;

  // Workspace: XO f32 [B][MP][C] (134MB) + two f16 x_tmp ping-pong buffers (67MB each)
  const size_t nE = (size_t)BB * MP * CC;
  float*     XO  = (float*)d_ws;
  _Float16*  XT0 = (_Float16*)((char*)d_ws + nE * sizeof(float));
  _Float16*  XT1 = XT0 + nE;

  crsae_init<<<2048, 256, 0, stream>>>(XO, XT0, nE);

  _Float16* xr = XT0;
  _Float16* xw = XT1;
  double s = 1.0;
  for (int t = 0; t < TT; ++t) {
    double sn = (1.0 + sqrt(1.0 + 4.0 * s * s)) * 0.5;
    float mom = (float)((s - 1.0) / sn);  // (s_old-1)/s_new
    s = sn;
    crsae_iter<<<dim3(MP / TM, BB), 256, 0, stream>>>(y, H, xr, XO, xw, mom);
    _Float16* tmp = xr; xr = xw; xw = tmp;
  }

  crsae_final<<<dim3(NN / TN, BB), 256, 0, stream>>>(H, XO, out);
}
